// DPSOG_pointcharge_309237645347
// MI455X (gfx1250) — matrix operand order repaired and checked by reading, not compiled, not run on hardware
//
#include <hip/hip_runtime.h>
#include <hip/hip_bf16.h>
#include <math.h>

// ---------------------------------------------------------------------------
// DP-SOG point-charge energy+forces, CDNA5 (gfx1250, wave32, WMMA bf16).
//
// Pair-stage MLPs: 16-pair x 16x16x32 bf16 WMMA tiles (f32 accumulate),
// K-masked A fragments, epilogue fused on the accumulator registers.
// Backward = WMMA VJP with activation recompute, dY fused into the A-frag
// gather. Transcendentals on the HW trans pipe (v_exp/v_rcp/v_rsq),
// branchless. Whole problem lives in L2 (192MB) -> compute bound.
// ---------------------------------------------------------------------------

constexpr int   kB = 4;
constexpr int   kN = 2048;
constexpr int   kM = 128;
constexpr float kL = 20.0f;
constexpr float kLog2e = 1.4426950408889634f;

typedef __attribute__((ext_vector_type(16))) __bf16 v16bf;
typedef __attribute__((ext_vector_type(8)))  float  v8f;

struct Params {
    // 6 pyramids: p1, p2, p3, p4, plr, fit ; 5 layers each ; (W, b)
    const float* W[6][5];
    const float* Bs[6][5];
    const float* linW;
    const float* linB;
    const float* beta;
    const float* wc;
};

// ---------------- fast transcendentals (branchless, HW trans pipe) ----------
__device__ __forceinline__ float fast_tanh(float x) {
    float xc = fminf(fmaxf(x, -15.0f), 15.0f);
    float e  = __builtin_amdgcn_exp2f(xc * (2.0f * kLog2e));
    return (e - 1.0f) * __builtin_amdgcn_rcpf(e + 1.0f);
}

// ---------------- WMMA fragment helpers (CDNA5 VGPR layouts) ----------------
// A (16x32 bf16): lane = row M (0..15, dup hi half), slot s -> K per ISA table.
// kmax masks the K padding (compile-time after unrolling -> loads elided).
__device__ __forceinline__ v16bf load_Afrag(const float (*X)[32], int lane,
                                            int kmax) {
    int m  = lane & 15;
    int hi = lane >> 4;
    v16bf a;
#pragma unroll
    for (int s = 0; s < 16; ++s) {
        int k = ((s < 8) ? s : (s + 8)) + 8 * hi;
        a[s] = (k < kmax) ? (__bf16)X[m][k] : (__bf16)0.0f;
    }
    return a;
}
// B (32x16 bf16): lane%16 = column N, slot s -> K per ISA table (same as A).
// W row-major [din][dout] f32 in LDS; zero-pad outside (din,dout).
__device__ __forceinline__ v16bf load_Bfrag(const float* W, int din, int dout,
                                            int col0, int lane) {
    int n  = (lane & 15) + col0;
    int kb = (lane >> 4) << 3;
    v16bf r;
#pragma unroll
    for (int s = 0; s < 16; ++s) {
        int   k = ((s < 8) ? s : (s + 8)) + kb;
        float v = (k < din && n < dout) ? W[k * dout + n] : 0.0f;
        r[s] = (__bf16)v;
    }
    return r;
}
// B = W^T (dout x din) for backward: elem(k,n) = W[n*dout+k].
__device__ __forceinline__ v16bf load_BTfrag(const float* W, int din, int dout,
                                             int lane) {
    int n  = lane & 15;
    int kb = (lane >> 4) << 3;
    v16bf r;
#pragma unroll
    for (int s = 0; s < 16; ++s) {
        int   k = ((s < 8) ? s : (s + 8)) + kb;
        float v = (k < dout && n < din) ? W[n * dout + k] : 0.0f;
        r[s] = (__bf16)v;
    }
    return r;
}

// packed-weight offset inside one pyramid blob (layers 0..4)
__device__ __forceinline__ int pyr_woff(int din0, int l) {
    int o = 0;
#pragma unroll
    for (int i = 0; i < 5; ++i)
        if (i < l) o += ((i == 0) ? din0 : (2 << (i - 1))) * (2 << i);
    return o;
}
__device__ __forceinline__ int pyr_boff(int l) { return (2 << l) - 2; }

// ---------------- per-tile descriptor features g (and cached geometry) ------
__device__ __forceinline__ void gen_coor_tile(
    const float* __restrict__ pos, const float* __restrict__ charge,
    const int* __restrict__ neigh, int b, int n, int t, int lane,
    float (*G)[8], float (*dxc)[3], float* r2c, float* qc, int* jc, float* mk) {
    if (lane < 16) {
        int   m  = t * 16 + lane;
        int   nb = neigh[(b * kN + n) * kM + m];
        int   j  = (nb < 0) ? 0 : nb;
        float xi0 = pos[(b * kN + n) * 3 + 0];
        float xi1 = pos[(b * kN + n) * 3 + 1];
        float xi2 = pos[(b * kN + n) * 3 + 2];
        float d0 = pos[(b * kN + j) * 3 + 0] - xi0;
        float d1 = pos[(b * kN + j) * 3 + 1] - xi1;
        float d2 = pos[(b * kN + j) * 3 + 2] - xi2;
        d0 -= kL * rintf(d0 * (1.0f / kL));
        d1 -= kL * rintf(d1 * (1.0f / kL));
        d2 -= kL * rintf(d2 * (1.0f / kL));
        float r2 = d0 * d0 + d1 * d1 + d2 * d2;
        float q  = charge[b * kN + j];
        bool  ok = (nb >= 0) && (r2 > 1e-8f);
        float r2v = ok ? r2 : 1.0f;
        float sv  = __builtin_amdgcn_rsqf(r2v);
        float iv  = __builtin_amdgcn_rcpf(r2v);
        float mv  = ok ? 1.0f : 0.0f;
        G[lane][0] = sv * mv;
        G[lane][1] = d0 * iv * mv;
        G[lane][2] = d1 * iv * mv;
        G[lane][3] = d2 * iv * mv;
        G[lane][4] = q * sv * mv;
        G[lane][5] = q * d0 * iv * mv;
        G[lane][6] = q * d1 * iv * mv;
        G[lane][7] = q * d2 * iv * mv;
        if (dxc) {
            dxc[lane][0] = d0; dxc[lane][1] = d1; dxc[lane][2] = d2;
            r2c[lane] = r2v; qc[lane] = q; jc[lane] = j; mk[lane] = mv;
        }
    }
}

// ---------------- 16-pair pyramid forward (saves every layer output) --------
// Bpy = zero-padded biases, 32 floats per layer.
__device__ __forceinline__ void pyr_forward_save(
    const float* Wpy, const float* Bpy, int din0, int gbase, int lane,
    const float (*G)[8], float (*X0)[32], float (*Out)[16][32]) {
    // X0: only din0 columns are ever read (A-frag is K-masked)
    for (int i = lane; i < 16 * din0; i += 32) {
        int m = i / din0, k = i - m * din0;
        X0[m][k] = G[m][gbase + k];
    }
    const float (*cur)[32] = (const float (*)[32])X0;
    const int n  = lane & 15;
    const int mb = (lane >> 4) << 3;
#pragma unroll
    for (int l = 0; l < 5; ++l) {
        int          din  = (l == 0) ? din0 : (2 << (l - 1));
        int          dout = 2 << l;
        const float* W    = Wpy + pyr_woff(din0, l);
        const float* Bb   = Bpy + l * 32;
        float (*nxt)[32]  = Out[l];

        v16bf a = load_Afrag(cur, lane, din);
        {   // fragment cols [0,16)
            v16bf b0 = load_Bfrag(W, din, dout, 0, lane);
            v8f   c0 = {};
            c0 = __builtin_amdgcn_wmma_f32_16x16x32_bf16(false, a, false, b0,
                                                         (short)0, c0, false, false);
            float bias = Bb[n];
#pragma unroll
            for (int r = 0; r < 8; ++r) {
                int   m = mb + r;
                float y = fast_tanh(c0[r] + bias);
                float v;
                if (dout == 2 * din)      v = y + cur[m][(n >= din) ? (n - din) : n];
                else if (dout == din)     v = y + cur[m][n];
                else                      v = y;
                nxt[m][n] = (n < dout) ? v : 0.0f;
            }
        }
        if (dout > 16) {   // fragment cols [16,32) : dout==32 -> always tile res
            v16bf b1 = load_Bfrag(W, din, dout, 16, lane);
            v8f   c1 = {};
            c1 = __builtin_amdgcn_wmma_f32_16x16x32_bf16(false, a, false, b1,
                                                         (short)0, c1, false, false);
            float bias = Bb[16 + n];
            int   nc   = 16 + n;
#pragma unroll
            for (int r = 0; r < 8; ++r) {
                int m = mb + r;
                nxt[m][nc] = fast_tanh(c1[r] + bias) + cur[m][nc - din];
            }
        }
        cur = (const float (*)[32])nxt;
    }
}

// ---------------- cooperative weight-pack load into LDS ---------------------
// Wp: packed f32 weights (4*688). Bp: zero-padded biases 4*(5*32).
__device__ __forceinline__ void load_pair_weights(const Params& P, float* Wp,
                                                  float* Bp) {
    for (int i = threadIdx.x; i < 4 * 160; i += 64) Bp[i] = 0.0f;
    __syncthreads();
#pragma unroll
    for (int py = 0; py < 4; ++py) {
        int din0 = (py & 1) ? 3 : 1;
#pragma unroll
        for (int l = 0; l < 5; ++l) {
            int          sz  = ((l == 0) ? din0 : (2 << (l - 1))) * (2 << l);
            const float* src = P.W[py][l];
            for (int i = threadIdx.x; i < sz; i += 64)
                Wp[py * 688 + pyr_woff(din0, l) + i] = src[i];
            const float* bs = P.Bs[py][l];
            for (int i = threadIdx.x; i < (2 << l); i += 64)
                Bp[py * 160 + l * 32 + i] = bs[i];
        }
    }
}

// ============================ kernel 1: pair forward =========================
__global__ __launch_bounds__(64) void pair_forward(
    const float* __restrict__ pos, const float* __restrict__ charge,
    const int* __restrict__ neigh, Params P, float* __restrict__ Dout) {
    __shared__ float Wp[4 * 688];
    __shared__ float Bp[4 * 160];
    __shared__ float X0s[2][16][32];
    __shared__ float Outs[2][5][16][32];
    __shared__ float Gs[2][16][8];
    __shared__ float Dacc[2][128];

    const int wv   = threadIdx.x >> 5;
    const int lane = threadIdx.x & 31;
    const int row  = blockIdx.x * 2 + wv;
    const int b    = row / kN;
    const int n    = row % kN;

    load_pair_weights(P, Wp, Bp);
    __syncthreads();

#pragma unroll
    for (int it = 0; it < 4; ++it) Dacc[wv][it * 32 + lane] = 0.0f;

    for (int t = 0; t < 8; ++t) {
        gen_coor_tile(pos, charge, neigh, b, n, t, lane, Gs[wv],
                      nullptr, nullptr, nullptr, nullptr, nullptr);
#pragma unroll
        for (int py = 0; py < 4; ++py) {
            int din0  = (py & 1) ? 3 : 1;
            int gbase = (py == 0) ? 0 : ((py == 1) ? 1 : ((py == 2) ? 4 : 5));
            pyr_forward_save(&Wp[py * 688], &Bp[py * 160], din0, gbase, lane,
                             Gs[wv], X0s[wv], Outs[wv]);
            // D[row, py*32+j] += sum_p s_p * out[p][j]
            {
                int   j   = lane;
                float acc = 0.0f;
#pragma unroll
                for (int p = 0; p < 16; ++p)
                    acc += Gs[wv][p][0] * Outs[wv][4][p][j];
                Dacc[wv][py * 32 + j] += acc;
            }
        }
    }
#pragma unroll
    for (int it = 0; it < 4; ++it)
        Dout[row * 128 + it * 32 + lane] = Dacc[wv][it * 32 + lane];
}

// ============================ kernel 2: sog forward ==========================
__global__ __launch_bounds__(256) void sog_forward(
    const float* __restrict__ pos, const float* __restrict__ charge,
    const float* __restrict__ beta, const float* __restrict__ wc,
    float* __restrict__ lrc) {
    __shared__ float pj[128][4];
    int   i  = blockIdx.x * 256 + threadIdx.x;
    int   b  = i / kN;
    int   n  = i % kN;
    // fold log2(e) into beta so the inner loop is one v_exp_f32 per channel
    float b0 = -beta[0] * kLog2e, b1 = -beta[1] * kLog2e;
    float b2 = -beta[2] * kLog2e, b3 = -beta[3] * kLog2e;
    float xi0 = pos[(b * kN + n) * 3 + 0];
    float xi1 = pos[(b * kN + n) * 3 + 1];
    float xi2 = pos[(b * kN + n) * 3 + 2];
    float a0 = 0, a1 = 0, a2 = 0, a3 = 0;
    for (int j0 = 0; j0 < kN; j0 += 128) {
        __syncthreads();
        if (threadIdx.x < 128) {
            int j = j0 + threadIdx.x;
            pj[threadIdx.x][0] = pos[(b * kN + j) * 3 + 0];
            pj[threadIdx.x][1] = pos[(b * kN + j) * 3 + 1];
            pj[threadIdx.x][2] = pos[(b * kN + j) * 3 + 2];
            pj[threadIdx.x][3] = charge[b * kN + j];
        }
        __syncthreads();
        for (int jj = 0; jj < 128; ++jj) {
            float d0 = xi0 - pj[jj][0]; d0 -= kL * rintf(d0 * (1.0f / kL));
            float d1 = xi1 - pj[jj][1]; d1 -= kL * rintf(d1 * (1.0f / kL));
            float d2 = xi2 - pj[jj][2]; d2 -= kL * rintf(d2 * (1.0f / kL));
            float r2 = d0 * d0 + d1 * d1 + d2 * d2;
            float q  = pj[jj][3];
            a0 += __builtin_amdgcn_exp2f(r2 * b0) * q;
            a1 += __builtin_amdgcn_exp2f(r2 * b1) * q;
            a2 += __builtin_amdgcn_exp2f(r2 * b2) * q;
            a3 += __builtin_amdgcn_exp2f(r2 * b3) * q;
        }
    }
    lrc[i * 4 + 0] = a0 * wc[0];
    lrc[i * 4 + 1] = a1 * wc[1];
    lrc[i * 4 + 2] = a2 * wc[2];
    lrc[i * 4 + 3] = a3 * wc[3];
}

// =================== kernel 3: row stage (fit/plr fwd + bwd, E) =============
__global__ __launch_bounds__(256) void row_stage(
    const float* __restrict__ Dw, const float* __restrict__ lrcw, Params P,
    float* __restrict__ dDw, float* __restrict__ dlrcw,
    float* __restrict__ Eout) {
    int i = blockIdx.x * 256 + threadIdx.x;
    int b = i / kN;

    // ---- plr forward (pyramid 4, din0=4) ----
    float x4[4];
#pragma unroll
    for (int k = 0; k < 4; ++k) x4[k] = lrcw[i * 4 + k];
    float o0[2], o1[4], o2[8], o3[16], o4[32];
    {   const float* W = P.W[4][0]; const float* Bb = P.Bs[4][0];   // 4->2 (no res)
        for (int j = 0; j < 2; ++j) { float a = Bb[j];
            for (int k = 0; k < 4; ++k) a += x4[k] * W[k * 2 + j];
            o0[j] = fast_tanh(a); } }
    {   const float* W = P.W[4][1]; const float* Bb = P.Bs[4][1];   // 2->4 tile
        for (int j = 0; j < 4; ++j) { float a = Bb[j];
            for (int k = 0; k < 2; ++k) a += o0[k] * W[k * 4 + j];
            o1[j] = fast_tanh(a) + o0[j & 1]; } }
    {   const float* W = P.W[4][2]; const float* Bb = P.Bs[4][2];   // 4->8 tile
        for (int j = 0; j < 8; ++j) { float a = Bb[j];
            for (int k = 0; k < 4; ++k) a += o1[k] * W[k * 8 + j];
            o2[j] = fast_tanh(a) + o1[j & 3]; } }
    {   const float* W = P.W[4][3]; const float* Bb = P.Bs[4][3];   // 8->16 tile
        for (int j = 0; j < 16; ++j) { float a = Bb[j];
            for (int k = 0; k < 8; ++k) a += o2[k] * W[k * 16 + j];
            o3[j] = fast_tanh(a) + o2[j & 7]; } }
    {   const float* W = P.W[4][4]; const float* Bb = P.Bs[4][4];   // 16->32 tile
        for (int j = 0; j < 32; ++j) { float a = Bb[j];
            for (int k = 0; k < 16; ++k) a += o3[k] * W[k * 32 + j];
            o4[j] = fast_tanh(a) + o3[j & 15]; } }

    // ---- fit forward (pyramid 5, 160->16->8->4->2->1, no residuals) ----
    float f0[16], f1[8], f2[4], f3[2], f4[1];
    {   const float* W = P.W[5][0]; const float* Bb = P.Bs[5][0];
        for (int j = 0; j < 16; ++j) { float a = Bb[j];
            for (int k = 0; k < 128; ++k) a += Dw[i * 128 + k] * W[k * 16 + j];
            for (int k = 0; k < 32; ++k)  a += o4[k] * W[(128 + k) * 16 + j];
            f0[j] = fast_tanh(a); } }
    {   const float* W = P.W[5][1]; const float* Bb = P.Bs[5][1];
        for (int j = 0; j < 8; ++j) { float a = Bb[j];
            for (int k = 0; k < 16; ++k) a += f0[k] * W[k * 8 + j];
            f1[j] = fast_tanh(a); } }
    {   const float* W = P.W[5][2]; const float* Bb = P.Bs[5][2];
        for (int j = 0; j < 4; ++j) { float a = Bb[j];
            for (int k = 0; k < 8; ++k) a += f1[k] * W[k * 4 + j];
            f2[j] = fast_tanh(a); } }
    {   const float* W = P.W[5][3]; const float* Bb = P.Bs[5][3];
        for (int j = 0; j < 2; ++j) { float a = Bb[j];
            for (int k = 0; k < 4; ++k) a += f2[k] * W[k * 2 + j];
            f3[j] = fast_tanh(a); } }
    {   const float* W = P.W[5][4]; const float* Bb = P.Bs[5][4];
        float a = Bb[0];
        for (int k = 0; k < 2; ++k) a += f3[k] * W[k];
        f4[0] = fast_tanh(a); }

    float F = f4[0] * P.linW[0] + P.linB[0];
    atomicAdd(&Eout[b], F);

    // ---- fit backward (dE/dF = 1) ----
    float dY4 = P.linW[0] * (1.0f - f4[0] * f4[0]);
    float d3f[2], d2f[4], d1f[8], d0f[16];
    {   const float* W = P.W[5][4];
        for (int k = 0; k < 2; ++k) d3f[k] = dY4 * W[k]; }
    {   const float* W = P.W[5][3];
        float dY[2]; for (int j = 0; j < 2; ++j) dY[j] = d3f[j] * (1.0f - f3[j] * f3[j]);
        for (int k = 0; k < 4; ++k) { float a = 0;
            for (int j = 0; j < 2; ++j) a += dY[j] * W[k * 2 + j];
            d2f[k] = a; } }
    {   const float* W = P.W[5][2];
        float dY[4]; for (int j = 0; j < 4; ++j) dY[j] = d2f[j] * (1.0f - f2[j] * f2[j]);
        for (int k = 0; k < 8; ++k) { float a = 0;
            for (int j = 0; j < 4; ++j) a += dY[j] * W[k * 4 + j];
            d1f[k] = a; } }
    {   const float* W = P.W[5][1];
        float dY[8]; for (int j = 0; j < 8; ++j) dY[j] = d1f[j] * (1.0f - f1[j] * f1[j]);
        for (int k = 0; k < 16; ++k) { float a = 0;
            for (int j = 0; j < 8; ++j) a += dY[j] * W[k * 8 + j];
            d0f[k] = a; } }
    float dL3[32];
    {   const float* W = P.W[5][0];
        float dY[16];
        for (int j = 0; j < 16; ++j) dY[j] = d0f[j] * (1.0f - f0[j] * f0[j]);
        for (int k = 0; k < 128; ++k) { float a = 0;
            for (int j = 0; j < 16; ++j) a += dY[j] * W[k * 16 + j];
            dDw[i * 128 + k] = a; }
        for (int k = 0; k < 32; ++k) { float a = 0;
            for (int j = 0; j < 16; ++j) a += dY[j] * W[(128 + k) * 16 + j];
            dL3[k] = a; } }

    // ---- plr backward (tile residuals) ----
    float d3p[16], d2p[8], d1p[4], d0p[2];
    {   const float* W = P.W[4][4];                                   // 16->32 tile
        float dY[32];
        for (int j = 0; j < 32; ++j) { float y = o4[j] - o3[j & 15];
            dY[j] = dL3[j] * (1.0f - y * y); }
        for (int k = 0; k < 16; ++k) { float a = dL3[k] + dL3[k + 16];
            for (int j = 0; j < 32; ++j) a += dY[j] * W[k * 32 + j];
            d3p[k] = a; } }
    {   const float* W = P.W[4][3];                                   // 8->16 tile
        float dY[16];
        for (int j = 0; j < 16; ++j) { float y = o3[j] - o2[j & 7];
            dY[j] = d3p[j] * (1.0f - y * y); }
        for (int k = 0; k < 8; ++k) { float a = d3p[k] + d3p[k + 8];
            for (int j = 0; j < 16; ++j) a += dY[j] * W[k * 16 + j];
            d2p[k] = a; } }
    {   const float* W = P.W[4][2];                                   // 4->8 tile
        float dY[8];
        for (int j = 0; j < 8; ++j) { float y = o2[j] - o1[j & 3];
            dY[j] = d2p[j] * (1.0f - y * y); }
        for (int k = 0; k < 4; ++k) { float a = d2p[k] + d2p[k + 4];
            for (int j = 0; j < 8; ++j) a += dY[j] * W[k * 8 + j];
            d1p[k] = a; } }
    {   const float* W = P.W[4][1];                                   // 2->4 tile
        float dY[4];
        for (int j = 0; j < 4; ++j) { float y = o1[j] - o0[j & 1];
            dY[j] = d1p[j] * (1.0f - y * y); }
        for (int k = 0; k < 2; ++k) { float a = d1p[k] + d1p[k + 2];
            for (int j = 0; j < 4; ++j) a += dY[j] * W[k * 4 + j];
            d0p[k] = a; } }
    {   const float* W = P.W[4][0];                                   // 4->2 no res
        float dY[2];
        for (int j = 0; j < 2; ++j) dY[j] = d0p[j] * (1.0f - o0[j] * o0[j]);
        for (int k = 0; k < 4; ++k) { float a = 0;
            for (int j = 0; j < 2; ++j) a += dY[j] * W[k * 2 + j];
            dlrcw[i * 4 + k] = a; } }
}

// ============================ kernel 4: pair backward ========================
__global__ __launch_bounds__(64) void pair_backward(
    const float* __restrict__ pos, const float* __restrict__ charge,
    const int* __restrict__ neigh, Params P, const float* __restrict__ dD,
    float* __restrict__ Fout) {
    __shared__ float Wp[4 * 688];
    __shared__ float Bp[4 * 160];
    __shared__ float X0s[2][16][32];
    __shared__ float Outs[2][5][16][32];
    __shared__ float dB1[2][16][32];
    __shared__ float dB2[2][16][32];
    __shared__ float Gs[2][16][8];
    __shared__ float dxs[2][16][3];
    __shared__ float r2s[2][16];
    __shared__ float qjs[2][16];
    __shared__ int   jis[2][16];
    __shared__ float mks[2][16];
    __shared__ float dgs[2][16][8];
    __shared__ float dsv[2][16];
    __shared__ float dDl[2][128];

    const int wv   = threadIdx.x >> 5;
    const int lane = threadIdx.x & 31;
    const int row  = blockIdx.x * 2 + wv;
    const int b    = row / kN;
    const int n    = row % kN;

    load_pair_weights(P, Wp, Bp);
    __syncthreads();

#pragma unroll
    for (int it = 0; it < 4; ++it)
        dDl[wv][it * 32 + lane] = dD[row * 128 + it * 32 + lane];

    for (int t = 0; t < 8; ++t) {
        gen_coor_tile(pos, charge, neigh, b, n, t, lane, Gs[wv],
                      dxs[wv], r2s[wv], qjs[wv], jis[wv], mks[wv]);
        if (lane < 16) dsv[wv][lane] = 0.0f;

#pragma unroll
        for (int py = 0; py < 4; ++py) {
            int din0  = (py & 1) ? 3 : 1;
            int gbase = (py == 0) ? 0 : ((py == 1) ? 1 : ((py == 2) ? 4 : 5));
            // recompute forward with all layer outputs
            pyr_forward_save(&Wp[py * 688], &Bp[py * 160], din0, gbase, lane,
                             Gs[wv], X0s[wv], Outs[wv]);
            // dE/ds contribution: sum_j out[j]*dD[py*32+j]
            if (lane < 16) {
                float a = 0.0f;
#pragma unroll
                for (int j = 0; j < 32; ++j)
                    a += Outs[wv][4][lane][j] * dDl[wv][py * 32 + j];
                dsv[wv][lane] += a;
            }
            // dCur = s * dD (broadcast over pairs of row)
#pragma unroll
            for (int it = 0; it < 16; ++it) {
                int i = it * 32 + lane;
                int m = i >> 5, j = i & 31;
                dB1[wv][m][j] = Gs[wv][m][0] * dDl[wv][py * 32 + j];
            }
            float (*pCur)[32] = dB1[wv];
            float (*pNxt)[32] = dB2[wv];
#pragma unroll
            for (int l = 4; l >= 0; --l) {
                int din  = (l == 0) ? din0 : (2 << (l - 1));
                int dout = 2 << l;
                const float (*xin)[32] =
                    (l == 0) ? (const float (*)[32])X0s[wv]
                             : (const float (*)[32])Outs[wv][l - 1];
                const float (*OutL)[32] = (const float (*)[32])Outs[wv][l];
                const float* W = &Wp[py * 688 + pyr_woff(din0, l)];
                // fused A-frag: dY = dOut * (1 - tanh^2), tanh = out - residual
                v16bf a;
                {
                    int mA = lane & 15, hi = lane >> 4;
#pragma unroll
                    for (int s = 0; s < 16; ++s) {
                        int   k = ((s < 8) ? s : (s + 8)) + 8 * hi;
                        float v = 0.0f;
                        if (k < dout) {
                            float res = (dout == 2 * din)
                                            ? xin[mA][(k >= din) ? (k - din) : k]
                                            : ((dout == din) ? xin[mA][k] : 0.0f);
                            float y = OutL[mA][k] - res;
                            v = pCur[mA][k] * (1.0f - y * y);
                        }
                        a[s] = (__bf16)v;
                    }
                }
                v16bf bf = load_BTfrag(W, din, dout, lane);
                v8f   c  = {};
                c = __builtin_amdgcn_wmma_f32_16x16x32_bf16(false, a, false, bf,
                                                            (short)0, c, false, false);
                // fragment scatter: new dCur = dY @ W^T + residual VJP
                {
                    int nn = lane & 15, mb = (lane >> 4) << 3;
#pragma unroll
                    for (int r = 0; r < 8; ++r) {
                        int   m = mb + r;
                        float v = c[r];
                        if (dout == 2 * din)      v += pCur[m][nn] + pCur[m][nn + din];
                        else if (dout == din)     v += pCur[m][nn];
                        pNxt[m][nn] = (nn < din) ? v : 0.0f;
                    }
                }
                float (*sw)[32] = pCur; pCur = pNxt; pNxt = sw;
            }
            if (lane < 16) {
#pragma unroll
                for (int k = 0; k < 3; ++k)
                    if (k < din0) dgs[wv][lane][gbase + k] = pCur[lane][k];
            }
        }

        // ---- gen_coor VJP -> forces (= -grad) ----
        if (lane < 16) {
            float mv = mks[wv][lane];
            if (mv != 0.0f) {
                float r2v = r2s[wv][lane];
                float iv  = __builtin_amdgcn_rcpf(r2v);
                float sv  = __builtin_amdgcn_rsqf(r2v);
                float s3  = sv * iv;
                float q   = qjs[wv][lane];
                float A0  = (dgs[wv][lane][0] + dsv[wv][lane]) + q * dgs[wv][lane][4];
                float w0  = dgs[wv][lane][1] + q * dgs[wv][lane][5];
                float w1  = dgs[wv][lane][2] + q * dgs[wv][lane][6];
                float w2  = dgs[wv][lane][3] + q * dgs[wv][lane][7];
                float d0 = dxs[wv][lane][0], d1 = dxs[wv][lane][1], d2 = dxs[wv][lane][2];
                float wd = w0 * d0 + w1 * d1 + w2 * d2;
                float g0 = -A0 * s3 * d0 + w0 * iv - 2.0f * wd * d0 * iv * iv;
                float g1 = -A0 * s3 * d1 + w1 * iv - 2.0f * wd * d1 * iv * iv;
                float g2 = -A0 * s3 * d2 + w2 * iv - 2.0f * wd * d2 * iv * iv;
                int   j  = jis[wv][lane];
                atomicAdd(&Fout[(b * kN + j) * 3 + 0], -g0);   // neighbor
                atomicAdd(&Fout[(b * kN + j) * 3 + 1], -g1);
                atomicAdd(&Fout[(b * kN + j) * 3 + 2], -g2);
                atomicAdd(&Fout[(b * kN + n) * 3 + 0], g0);    // center
                atomicAdd(&Fout[(b * kN + n) * 3 + 1], g1);
                atomicAdd(&Fout[(b * kN + n) * 3 + 2], g2);
            }
        }
    }
}

// ============================ kernel 5: sog backward =========================
__global__ __launch_bounds__(256) void sog_backward(
    const float* __restrict__ pos, const float* __restrict__ charge,
    const float* __restrict__ beta, const float* __restrict__ wc,
    const float* __restrict__ dlrcw, float* __restrict__ Fout) {
    __shared__ float pj[128][8];  // x,y,z,q,dl0..dl3
    int   i  = blockIdx.x * 256 + threadIdx.x;
    int   b  = i / kN;
    int   n  = i % kN;
    float bb0 = beta[0], bb1 = beta[1], bb2 = beta[2], bb3 = beta[3];
    float e0 = -bb0 * kLog2e, e1 = -bb1 * kLog2e;   // exp2 scale
    float e2 = -bb2 * kLog2e, e3 = -bb3 * kLog2e;
    float w0 = 2.0f * wc[0] * bb0, w1 = 2.0f * wc[1] * bb1;
    float w2 = 2.0f * wc[2] * bb2, w3 = 2.0f * wc[3] * bb3;
    float xi0 = pos[(b * kN + n) * 3 + 0];
    float xi1 = pos[(b * kN + n) * 3 + 1];
    float xi2 = pos[(b * kN + n) * 3 + 2];
    float qi  = charge[b * kN + n];
    float di0 = dlrcw[i * 4 + 0], di1 = dlrcw[i * 4 + 1];
    float di2 = dlrcw[i * 4 + 2], di3 = dlrcw[i * 4 + 3];
    float f0 = 0, f1 = 0, f2 = 0;
    for (int j0 = 0; j0 < kN; j0 += 128) {
        __syncthreads();
        if (threadIdx.x < 128) {
            int j = j0 + threadIdx.x;
            pj[threadIdx.x][0] = pos[(b * kN + j) * 3 + 0];
            pj[threadIdx.x][1] = pos[(b * kN + j) * 3 + 1];
            pj[threadIdx.x][2] = pos[(b * kN + j) * 3 + 2];
            pj[threadIdx.x][3] = charge[b * kN + j];
            pj[threadIdx.x][4] = dlrcw[(b * kN + j) * 4 + 0];
            pj[threadIdx.x][5] = dlrcw[(b * kN + j) * 4 + 1];
            pj[threadIdx.x][6] = dlrcw[(b * kN + j) * 4 + 2];
            pj[threadIdx.x][7] = dlrcw[(b * kN + j) * 4 + 3];
        }
        __syncthreads();
        for (int jj = 0; jj < 128; ++jj) {
            float d0 = xi0 - pj[jj][0]; d0 -= kL * rintf(d0 * (1.0f / kL));
            float d1 = xi1 - pj[jj][1]; d1 -= kL * rintf(d1 * (1.0f / kL));
            float d2 = xi2 - pj[jj][2]; d2 -= kL * rintf(d2 * (1.0f / kL));
            float r2 = d0 * d0 + d1 * d1 + d2 * d2;
            float qj = pj[jj][3];
            float s  = (di0 * qj + pj[jj][4] * qi) * w0 * __builtin_amdgcn_exp2f(r2 * e0) +
                       (di1 * qj + pj[jj][5] * qi) * w1 * __builtin_amdgcn_exp2f(r2 * e1) +
                       (di2 * qj + pj[jj][6] * qi) * w2 * __builtin_amdgcn_exp2f(r2 * e2) +
                       (di3 * qj + pj[jj][7] * qi) * w3 * __builtin_amdgcn_exp2f(r2 * e3);
            f0 += s * d0; f1 += s * d1; f2 += s * d2;
        }
    }
    atomicAdd(&Fout[i * 3 + 0], f0);
    atomicAdd(&Fout[i * 3 + 1], f1);
    atomicAdd(&Fout[i * 3 + 2], f2);
}

// ================================ launcher ===================================
extern "C" void kernel_launch(void* const* d_in, const int* in_sizes, int n_in,
                              void* d_out, int out_size, void* d_ws,
                              size_t ws_size, hipStream_t stream) {
    (void)in_sizes; (void)n_in; (void)ws_size;
    const float* pos    = (const float*)d_in[0];
    const float* charge = (const float*)d_in[1];
    const int*   neigh  = (const int*)d_in[2];

    // params assumed flattened in setup_inputs() insertion order:
    // p1..p4, plr, fit (each: 5 x (W,b)), lin(W,b), beta, wc
    Params P;
    int idx = 3;
    for (int py = 0; py < 6; ++py)
        for (int l = 0; l < 5; ++l) {
            P.W[py][l]  = (const float*)d_in[idx++];
            P.Bs[py][l] = (const float*)d_in[idx++];
        }
    P.linW = (const float*)d_in[idx++];
    P.linB = (const float*)d_in[idx++];
    P.beta = (const float*)d_in[idx++];
    P.wc   = (const float*)d_in[idx++];

    float* ws    = (float*)d_ws;
    float* Dw    = ws;                       // 8192*128
    float* dDw   = Dw + kB * kN * 128;       // 8192*128
    float* lrcw  = dDw + kB * kN * 128;      // 8192*4
    float* dlrcw = lrcw + kB * kN * 4;       // 8192*4

    float* E = (float*)d_out;                // [B]   energies
    float* F = E + kB;                       // [B*N*3] forces

    hipMemsetAsync(d_out, 0, (size_t)out_size * sizeof(float), stream);

    pair_forward<<<kB * kN / 2, 64, 0, stream>>>(pos, charge, neigh, P, Dw);
    sog_forward<<<kB * kN / 256, 256, 0, stream>>>(pos, charge, P.beta, P.wc, lrcw);
    row_stage<<<kB * kN / 256, 256, 0, stream>>>(Dw, lrcw, P, dDw, dlrcw, E);
    pair_backward<<<kB * kN / 2, 64, 0, stream>>>(pos, charge, neigh, P, dDw, F);
    sog_backward<<<kB * kN / 256, 256, 0, stream>>>(pos, charge, P.beta, P.wc,
                                                    dlrcw, F);
}
